// TalkingHeadAttention_42812234006693
// MI455X (gfx1250) — matrix operand order repaired and checked by reading, not compiled, not run on hardware
//
#include <hip/hip_runtime.h>

#define B_ 8
#define N_ 1024
#define C_ 768
#define H_ 12
#define D_ 64
#define SCALE_ 0.125f   // 64^-0.5

typedef __attribute__((ext_vector_type(16))) __bf16 v16bf;
typedef __attribute__((ext_vector_type(8)))  float  v8f;
typedef __attribute__((ext_vector_type(4)))  unsigned int u32x4;
typedef __attribute__((ext_vector_type(8)))  unsigned int u32x8;
typedef __attribute__((ext_vector_type(4)))  float  f32x4;

// ---------------------------------------------------------------------------
// WMMA fragment helpers (gfx1250 wave32, V_WMMA_F32_16X16X32_BF16)
// A-matrix 16x32 (MxK): lanes 0-15 -> M=lane, K={0..7,16..23};
//                       lanes 16-31 -> M=lane-16, K={8..15,24..31}
// B-matrix 32x16 (KxN): lane 0-15 -> N=lane, K={0..7,16..23}; lanes 16-31 -> K={8..15,24..31}
// C/D 16x16 f32: VGPR r: lanes 0-15 -> M=r, N=lane; lanes 16-31 -> M=8+r
// ---------------------------------------------------------------------------

static __device__ inline v16bf load_fragA_bf16(const __bf16* base, int ld, int lane) {
  const int r  = lane & 15;
  const int kb = (lane & 16) ? 8 : 0;
  const __bf16* p = base + (size_t)r * ld + kb;
  union { v16bf v; u32x4 u[2]; } f;
  f.u[0] = *reinterpret_cast<const u32x4*>(p);        // K = kb .. kb+7
  f.u[1] = *reinterpret_cast<const u32x4*>(p + 16);   // K = kb+16 .. kb+23
  return f.v;
}

static __device__ inline v16bf load_fragA_f32(const float* base, int ld, int lane) {
  const int r  = lane & 15;
  const int kb = (lane & 16) ? 8 : 0;
  const float* p = base + (size_t)r * ld + kb;
  f32x4 a0 = *reinterpret_cast<const f32x4*>(p);
  f32x4 a1 = *reinterpret_cast<const f32x4*>(p + 4);
  f32x4 a2 = *reinterpret_cast<const f32x4*>(p + 16);
  f32x4 a3 = *reinterpret_cast<const f32x4*>(p + 20);
  v16bf v;
  v[0]  = (__bf16)a0.x; v[1]  = (__bf16)a0.y; v[2]  = (__bf16)a0.z; v[3]  = (__bf16)a0.w;
  v[4]  = (__bf16)a1.x; v[5]  = (__bf16)a1.y; v[6]  = (__bf16)a1.z; v[7]  = (__bf16)a1.w;
  v[8]  = (__bf16)a2.x; v[9]  = (__bf16)a2.y; v[10] = (__bf16)a2.z; v[11] = (__bf16)a2.w;
  v[12] = (__bf16)a3.x; v[13] = (__bf16)a3.y; v[14] = (__bf16)a3.z; v[15] = (__bf16)a3.w;
  return v;
}

// bt_base points at row n0 of B^T (rows of B^T are columns of B), row length ld
static __device__ inline v16bf load_fragB_bf16(const __bf16* bt_base, int ld, int lane) {
  const int c  = lane & 15;
  const int kb = (lane & 16) ? 8 : 0;
  const __bf16* p = bt_base + (size_t)c * ld + kb;
  union { v16bf v; u32x4 u[2]; } f;
  f.u[0] = *reinterpret_cast<const u32x4*>(p);        // K = kb .. kb+7
  f.u[1] = *reinterpret_cast<const u32x4*>(p + 16);   // K = kb+16 .. kb+23
  return f.v;
}

static __device__ inline v8f wmma_bf16(v16bf a, v16bf b, v8f c) {
  return __builtin_amdgcn_wmma_f32_16x16x32_bf16(false, a, false, b, (short)0, c, false, false);
}

// gfx1250 async copy: 16 bytes/lane global -> LDS, tracked on ASYNCcnt
static __device__ inline void async_copy16(unsigned lds_off, const void* gptr) {
  asm volatile("global_load_async_to_lds_b128 %0, %1, off"
               :: "v"(lds_off), "v"(gptr)
               : "memory");
}
static __device__ inline void wait_async0() {
  asm volatile("s_wait_asynccnt 0x0" ::: "memory");
}

// ---------------------------------------------------------------------------
// gfx1250 Tensor Data Mover: 2D tile load (rows x 64 bf16) global -> LDS,
// LDS rows padded +16B (pad_interval = 32 DWORDs, pad_amount = 4 DWORDs) to
// give a 144 B LDS row stride. D# groups per CDNA5 ISA §8.3/8.4; 2D tensor
// -> only groups 0 and 1 needed (VADDR2/3 omitted). Tracked on TENSORcnt.
// ---------------------------------------------------------------------------
static __device__ inline void tdm_load_tile_128x64(unsigned lds_off, const void* gaddr) {
  const unsigned long long ga = (unsigned long long)(uintptr_t)gaddr;
  u32x4 g0;
  g0[0] = 1u;                                           // count=1, user D#
  g0[1] = lds_off;                                      // lds_addr (bytes)
  g0[2] = (unsigned)ga;                                 // global_addr[31:0]
  g0[3] = (unsigned)((ga >> 32) & 0x01FFFFFFu) | (2u << 30);  // addr[56:32], type=2
  u32x8 g1;
  g1[0] = (1u << 16)            // data_size = 1 -> 2 bytes
        | (1u << 20)            // pad_enable
        | (4u << 22)            // pad_interval code 4 -> every 32 DWORDs (128 B row)
        | (3u << 25);           // pad_amount code 3 -> 4 DWORDs (16 B)
  g1[1] = (64u << 16);          // tensor_dim0 = 64 (bits 79:48, low half)
  g1[2] = (1024u << 16);        // tensor_dim0 hi = 0; tensor_dim1 lo = 1024
  g1[3] = (64u << 16);          // tensor_dim1 hi = 0; tile_dim0 = 64
  g1[4] = 128u;                 // tile_dim1 = 128; tile_dim2 = 0
  g1[5] = 64u;                  // tensor_dim0_stride = 64 (low 32)
  g1[6] = 0u;                   // stride hi, tensor_dim1_stride lo
  g1[7] = 0u;
  asm volatile("tensor_load_to_lds %0, %1"
               :: "s"(g0), "s"(g1)
               : "memory");
}

// ---------------------------------------------------------------------------
// Conversion kernels
// ---------------------------------------------------------------------------
__global__ __launch_bounds__(256) void cvt_f32_bf16(const float* __restrict__ in,
                                                    __bf16* __restrict__ out, size_t n) {
  for (size_t i = blockIdx.x * (size_t)blockDim.x + threadIdx.x; i < n;
       i += (size_t)gridDim.x * blockDim.x)
    out[i] = (__bf16)in[i];
}

// out is [cols][rows] bf16 (i.e. transpose of in[rows][cols])
__global__ __launch_bounds__(256) void transpose_f32_bf16(const float* __restrict__ in,
                                                          __bf16* __restrict__ out,
                                                          int rows, int cols) {
  size_t total = (size_t)rows * cols;
  for (size_t o = blockIdx.x * (size_t)blockDim.x + threadIdx.x; o < total;
       o += (size_t)gridDim.x * blockDim.x) {
    size_t r = o % rows, c = o / rows;
    out[o] = (__bf16)in[r * cols + c];
  }
}

// ---------------------------------------------------------------------------
// QKV projection: [8192 x 768] @ [768 x 2304] + bias; scatter into Q,K (BHND)
// and V^T (BHDN), all bf16. Q scaled by SCALE. 2-stage pipeline, unroll-2 so
// cur/next fragments ping-pong between register sets (no copy churn).
// ---------------------------------------------------------------------------
__global__ __launch_bounds__(256) void qkv_gemm(const __bf16* __restrict__ X,
                                                const __bf16* __restrict__ WT,
                                                const float* __restrict__ bqkv,
                                                __bf16* __restrict__ Qw,
                                                __bf16* __restrict__ Kw,
                                                __bf16* __restrict__ Vt) {
  const int lane = threadIdx.x & 31;
  const int wave = blockIdx.x * (blockDim.x >> 5) + (threadIdx.x >> 5);
  const int NB = (3 * C_) / 64;                 // 36 blocks of 64 cols
  const int mt = wave / NB, nb = wave % NB;
  if (mt >= (B_ * N_) / 16) return;
  const int m0 = mt * 16, n0 = nb * 64;

  const __bf16* Arow = X + (size_t)m0 * C_;
  v8f acc[4] = {};
  v16bf a_cur = load_fragA_bf16(Arow, C_, lane);
  v16bf b_cur[4];
#pragma unroll
  for (int j = 0; j < 4; ++j)
    b_cur[j] = load_fragB_bf16(WT + (size_t)(n0 + j * 16) * C_, C_, lane);

  const int steps = C_ / 32;                    // 24
#pragma unroll 2
  for (int s = 0; s < steps - 1; ++s) {
    const int kn = (s + 1) * 32;
    v16bf a_nxt = load_fragA_bf16(Arow + kn, C_, lane);
    v16bf b_nxt[4];
#pragma unroll
    for (int j = 0; j < 4; ++j)
      b_nxt[j] = load_fragB_bf16(WT + (size_t)(n0 + j * 16) * C_ + kn, C_, lane);
#pragma unroll
    for (int j = 0; j < 4; ++j) acc[j] = wmma_bf16(a_cur, b_cur[j], acc[j]);
    a_cur = a_nxt;
#pragma unroll
    for (int j = 0; j < 4; ++j) b_cur[j] = b_nxt[j];
  }
#pragma unroll
  for (int j = 0; j < 4; ++j) acc[j] = wmma_bf16(a_cur, b_cur[j], acc[j]);

  const int cn = lane & 15;
  const int mo = (lane & 16) ? 8 : 0;
#pragma unroll
  for (int j = 0; j < 4; ++j) {
    const int col = n0 + j * 16 + cn;
    const int t = col / C_;
    const int rem = col % C_;
    const int h = rem / D_, d = rem % D_;
    const float bias = bqkv[col];
#pragma unroll
    for (int r = 0; r < 8; ++r) {
      const int tok = m0 + mo + r;
      const int bb = tok >> 10, n = tok & (N_ - 1);
      const float val = acc[j][r] + bias;
      const size_t bh = (size_t)bb * H_ + h;
      if (t == 0)       Qw[(bh * N_ + n) * D_ + d] = (__bf16)(val * SCALE_);
      else if (t == 1)  Kw[(bh * N_ + n) * D_ + d] = (__bf16)val;
      else              Vt[(bh * D_ + d) * N_ + n] = (__bf16)val;
    }
  }
}

// ---------------------------------------------------------------------------
// Scores: per (b,h)  S = Qs @ K^T, 128x128 tile per workgroup.
// Q tile staged by the Tensor Data Mover (TENSORcnt, wave 0 issues),
// K tile staged by per-lane async copies (ASYNCcnt). LDS rows padded to
// 72 bf16 (144 B) to spread banks. Output fp32 into the attn region of d_out.
// ---------------------------------------------------------------------------
#define SC_LDS_LD 72   // padded row stride in bf16 elements (144 B)

__global__ __launch_bounds__(256) void scores_gemm(const __bf16* __restrict__ Qw,
                                                   const __bf16* __restrict__ Kw,
                                                   float* __restrict__ S) {
  __shared__ __bf16 Qs[128 * SC_LDS_LD];   // 18 KB
  __shared__ __bf16 Ks[128 * SC_LDS_LD];   // 18 KB

  const int t = threadIdx.x;
  const int lane = t & 31;
  const int w = t >> 5;                       // wave 0..7

  const int per_bh = (N_ / 128) * (N_ / 128); // 64
  const int bh = blockIdx.x / per_bh;
  const int rem = blockIdx.x % per_bh;
  const int mtile = rem / (N_ / 128);
  const int ntile = rem % (N_ / 128);
  const int m0 = mtile * 128, n0 = ntile * 128;

  const __bf16* Qb = Qw + ((size_t)bh * N_ + m0) * D_;  // contiguous 128x64 tile
  const __bf16* Kb = Kw + ((size_t)bh * N_ + n0) * D_;

  // K tile: 1024 chunks of 16 B via async-to-LDS, 4 per thread
  {
    const unsigned k_lds = (unsigned)(uintptr_t)Ks;
#pragma unroll
    for (int i = 0; i < 4; ++i) {
      const int chunk = t + 256 * i;          // 0..1023
      const int row = chunk >> 3, ci = chunk & 7;
      async_copy16(k_lds + (unsigned)(row * SC_LDS_LD * 2 + ci * 16),
                   Kb + row * D_ + ci * 8);
    }
  }
  // Q tile: one TDM descriptor issued by wave 0 (TDM ignores EXEC; 1 issue)
  if (w == 0) {
    tdm_load_tile_128x64((unsigned)(uintptr_t)Qs, Qb);
    __builtin_amdgcn_s_wait_tensorcnt(0);
  }
  wait_async0();
  __syncthreads();

  // wave w computes rows [m0 + w*16, +16) x all 128 cols
  v8f acc[8] = {};
#pragma unroll
  for (int k0 = 0; k0 < D_; k0 += 32) {
    v16bf a = load_fragA_bf16(Qs + (w * 16) * SC_LDS_LD + k0, SC_LDS_LD, lane);
#pragma unroll
    for (int j = 0; j < 8; ++j) {
      v16bf b = load_fragB_bf16(Ks + (j * 16) * SC_LDS_LD + k0, SC_LDS_LD, lane);
      acc[j] = wmma_bf16(a, b, acc[j]);
    }
  }

  float* Sb = S + (size_t)bh * N_ * N_;
  const int cn = lane & 15;
  const int mo = (lane & 16) ? 8 : 0;
  const int rbase = m0 + w * 16 + mo;
#pragma unroll
  for (int j = 0; j < 8; ++j)
#pragma unroll
    for (int r = 0; r < 8; ++r)
      Sb[(size_t)(rbase + r) * N_ + n0 + j * 16 + cn] = acc[j][r];
}

// ---------------------------------------------------------------------------
// Fused talking-heads: pre-mix (Wl,bl) -> softmax(m) -> post-mix (Ww,bw).
// One block per (b,n); reads S[b,:,n,:] (12x1024) into LDS, writes final
// attn[b,:,n,:] back over the same memory. Disjoint across blocks -> safe.
// ---------------------------------------------------------------------------
__global__ __launch_bounds__(256) void mix_softmax(float* __restrict__ attn,
                                                   const float* __restrict__ Wl,
                                                   const float* __restrict__ bl,
                                                   const float* __restrict__ Ww,
                                                   const float* __restrict__ bw) {
  __shared__ float buf[H_ * N_];        // 48 KB
  __shared__ float red[H_ * 256];       // 12 KB
  __shared__ float wl_s[H_ * H_], ww_s[H_ * H_], bl_s[H_], bw_s[H_];

  const int t = threadIdx.x;
  const int bn = blockIdx.x;
  const int b = bn >> 10, n = bn & (N_ - 1);

  if (t < H_ * H_) { wl_s[t] = Wl[t]; ww_s[t] = Ww[t]; }
  if (t < H_)      { bl_s[t] = bl[t]; bw_s[t] = bw[t]; }

  for (int h = 0; h < H_; ++h) {
    const float* row = attn + (((size_t)b * H_ + h) * N_ + n) * N_;
#pragma unroll
    for (int q = 0; q < 4; ++q) buf[h * N_ + t + 256 * q] = row[t + 256 * q];
  }
  __syncthreads();

  // pre-softmax head mix: T[g][m] = bl[g] + sum_h S[h][m] * Wl[h][g]
  float T[H_][4];
#pragma unroll
  for (int q = 0; q < 4; ++q) {
    const int m = t + 256 * q;
    float s[H_];
#pragma unroll
    for (int h = 0; h < H_; ++h) s[h] = buf[h * N_ + m];
#pragma unroll
    for (int g = 0; g < H_; ++g) {
      float a = bl_s[g];
#pragma unroll
      for (int h = 0; h < H_; ++h) a += s[h] * wl_s[h * H_ + g];
      T[g][q] = a;
    }
  }

  // row max over m
  float lmax[H_];
#pragma unroll
  for (int g = 0; g < H_; ++g) {
    float mx = T[g][0];
#pragma unroll
    for (int q = 1; q < 4; ++q) mx = fmaxf(mx, T[g][q]);
    lmax[g] = mx;
  }
#pragma unroll
  for (int g = 0; g < H_; ++g) red[g * 256 + t] = lmax[g];
  __syncthreads();
  for (int st = 128; st > 0; st >>= 1) {
    if (t < st)
#pragma unroll
      for (int g = 0; g < H_; ++g)
        red[g * 256 + t] = fmaxf(red[g * 256 + t], red[g * 256 + t + st]);
    __syncthreads();
  }
  float rmax[H_];
#pragma unroll
  for (int g = 0; g < H_; ++g) rmax[g] = red[g * 256];
  __syncthreads();

  // exp + row sum
  float lsum[H_];
#pragma unroll
  for (int g = 0; g < H_; ++g) lsum[g] = 0.f;
#pragma unroll
  for (int q = 0; q < 4; ++q)
#pragma unroll
    for (int g = 0; g < H_; ++g) {
      T[g][q] = __expf(T[g][q] - rmax[g]);
      lsum[g] += T[g][q];
    }
#pragma unroll
  for (int g = 0; g < H_; ++g) red[g * 256 + t] = lsum[g];
  __syncthreads();
  for (int st = 128; st > 0; st >>= 1) {
    if (t < st)
#pragma unroll
      for (int g = 0; g < H_; ++g)
        red[g * 256 + t] += red[g * 256 + t + st];
    __syncthreads();
  }
  float rinv[H_];
#pragma unroll
  for (int g = 0; g < H_; ++g) rinv[g] = 1.0f / red[g * 256];

  // write normalized probabilities back to buf
#pragma unroll
  for (int q = 0; q < 4; ++q) {
    const int m = t + 256 * q;
#pragma unroll
    for (int g = 0; g < H_; ++g) buf[g * N_ + m] = T[g][q] * rinv[g];
  }
  __syncthreads();

  // post-softmax head mix, write final attn
#pragma unroll
  for (int q = 0; q < 4; ++q) {
    const int m = t + 256 * q;
    float p[H_];
#pragma unroll
    for (int h = 0; h < H_; ++h) p[h] = buf[h * N_ + m];
#pragma unroll
    for (int g = 0; g < H_; ++g) {
      float a = bw_s[g];
#pragma unroll
      for (int h = 0; h < H_; ++h) a += p[h] * ww_s[h * H_ + g];
      attn[(((size_t)b * H_ + g) * N_ + n) * N_ + m] = a;
    }
  }
}

// ---------------------------------------------------------------------------
// AV: per (b,h)  OH_head = attn[1024x1024 fp32] @ V[1024x64]; attn converted
// to bf16 in-register. 2-stage pipeline (unroll-2). Merged heads -> OH bf16.
// ---------------------------------------------------------------------------
__global__ __launch_bounds__(256) void av_gemm(const float* __restrict__ A,
                                               const __bf16* __restrict__ Vt,
                                               __bf16* __restrict__ OH) {
  const int lane = threadIdx.x & 31;
  const int wave = blockIdx.x * (blockDim.x >> 5) + (threadIdx.x >> 5);
  const int bh = wave / (N_ / 16);
  if (bh >= B_ * H_) return;
  const int mt = wave % (N_ / 16);
  const int m0 = mt * 16;

  const float* Ab = A + (size_t)bh * N_ * N_ + (size_t)m0 * N_;
  const __bf16* Vb = Vt + (size_t)bh * D_ * N_;   // rows of V^T = columns of V

  v8f acc[4] = {};
  v16bf a_cur = load_fragA_f32(Ab, N_, lane);
  v16bf b_cur[4];
#pragma unroll
  for (int j = 0; j < 4; ++j)
    b_cur[j] = load_fragB_bf16(Vb + (size_t)(j * 16) * N_, N_, lane);

  const int steps = N_ / 32;                      // 32
#pragma unroll 2
  for (int s = 0; s < steps - 1; ++s) {
    const int kn = (s + 1) * 32;
    v16bf a_nxt = load_fragA_f32(Ab + kn, N_, lane);
    v16bf b_nxt[4];
#pragma unroll
    for (int j = 0; j < 4; ++j)
      b_nxt[j] = load_fragB_bf16(Vb + (size_t)(j * 16) * N_ + kn, N_, lane);
#pragma unroll
    for (int j = 0; j < 4; ++j) acc[j] = wmma_bf16(a_cur, b_cur[j], acc[j]);
    a_cur = a_nxt;
#pragma unroll
    for (int j = 0; j < 4; ++j) b_cur[j] = b_nxt[j];
  }
#pragma unroll
  for (int j = 0; j < 4; ++j) acc[j] = wmma_bf16(a_cur, b_cur[j], acc[j]);

  const int b_ = bh / H_, h = bh % H_;
  const int cn = lane & 15;
  const int mo = (lane & 16) ? 8 : 0;
#pragma unroll
  for (int j = 0; j < 4; ++j)
#pragma unroll
    for (int r = 0; r < 8; ++r)
      OH[((size_t)b_ * N_ + m0 + mo + r) * C_ + h * D_ + j * 16 + cn] =
          (__bf16)acc[j][r];
}

// ---------------------------------------------------------------------------
// Output projection: [8192 x 768] @ Wp[768 x 768] + bp -> fp32 d_out.
// ---------------------------------------------------------------------------
__global__ __launch_bounds__(256) void proj_gemm(const __bf16* __restrict__ OH,
                                                 const __bf16* __restrict__ WpT,
                                                 const float* __restrict__ bp,
                                                 float* __restrict__ Out) {
  const int lane = threadIdx.x & 31;
  const int wave = blockIdx.x * (blockDim.x >> 5) + (threadIdx.x >> 5);
  const int NB = C_ / 64;                       // 12
  const int mt = wave / NB, nb = wave % NB;
  if (mt >= (B_ * N_) / 16) return;
  const int m0 = mt * 16, n0 = nb * 64;

  const __bf16* Arow = OH + (size_t)m0 * C_;
  v8f acc[4] = {};
  v16bf a_cur = load_fragA_bf16(Arow, C_, lane);
  v16bf b_cur[4];
#pragma unroll
  for (int j = 0; j < 4; ++j)
    b_cur[j] = load_fragB_bf16(WpT + (size_t)(n0 + j * 16) * C_, C_, lane);

  const int steps = C_ / 32;                    // 24
#pragma unroll 2
  for (int s = 0; s < steps - 1; ++s) {
    const int kn = (s + 1) * 32;
    v16bf a_nxt = load_fragA_bf16(Arow + kn, C_, lane);
    v16bf b_nxt[4];
#pragma unroll
    for (int j = 0; j < 4; ++j)
      b_nxt[j] = load_fragB_bf16(WpT + (size_t)(n0 + j * 16) * C_ + kn, C_, lane);
#pragma unroll
    for (int j = 0; j < 4; ++j) acc[j] = wmma_bf16(a_cur, b_cur[j], acc[j]);
    a_cur = a_nxt;
#pragma unroll
    for (int j = 0; j < 4; ++j) b_cur[j] = b_nxt[j];
  }
#pragma unroll
  for (int j = 0; j < 4; ++j) acc[j] = wmma_bf16(a_cur, b_cur[j], acc[j]);

  const int cn = lane & 15;
  const int mo = (lane & 16) ? 8 : 0;
#pragma unroll
  for (int j = 0; j < 4; ++j) {
    const int col = n0 + j * 16 + cn;
    const float bias = bp[col];
#pragma unroll
    for (int r = 0; r < 8; ++r)
      Out[(size_t)(m0 + mo + r) * C_ + col] = acc[j][r] + bias;
  }
}

// ---------------------------------------------------------------------------
extern "C" void kernel_launch(void* const* d_in, const int* in_sizes, int n_in,
                              void* d_out, int out_size, void* d_ws, size_t ws_size,
                              hipStream_t stream) {
  (void)in_sizes; (void)n_in; (void)out_size; (void)ws_size;
  const float* x    = (const float*)d_in[0];
  const float* Wqkv = (const float*)d_in[1];
  const float* bqkv = (const float*)d_in[2];
  const float* Wl   = (const float*)d_in[3];
  const float* bl   = (const float*)d_in[4];
  const float* Ww   = (const float*)d_in[5];
  const float* bw   = (const float*)d_in[6];
  const float* Wp   = (const float*)d_in[7];
  const float* bp   = (const float*)d_in[8];

  float* out  = (float*)d_out;
  float* attn = out + (size_t)B_ * N_ * C_;     // second tuple output region

  char* w = (char*)d_ws;
  auto carve = [&](size_t bytes) {
    char* p = w;
    w += (bytes + 255) & ~(size_t)255;
    return p;
  };
  __bf16* xb    = (__bf16*)carve((size_t)B_ * N_ * C_ * 2);
  __bf16* WqkvT = (__bf16*)carve((size_t)3 * C_ * C_ * 2);
  __bf16* WpT   = (__bf16*)carve((size_t)C_ * C_ * 2);
  __bf16* Qw    = (__bf16*)carve((size_t)B_ * H_ * N_ * D_ * 2);
  __bf16* Kw    = (__bf16*)carve((size_t)B_ * H_ * N_ * D_ * 2);
  __bf16* Vt    = (__bf16*)carve((size_t)B_ * H_ * D_ * N_ * 2);
  __bf16* OH    = (__bf16*)carve((size_t)B_ * N_ * C_ * 2);

  cvt_f32_bf16<<<2048, 256, 0, stream>>>(x, xb, (size_t)B_ * N_ * C_);
  transpose_f32_bf16<<<1024, 256, 0, stream>>>(Wqkv, WqkvT, C_, 3 * C_);
  transpose_f32_bf16<<<512, 256, 0, stream>>>(Wp, WpT, C_, C_);

  qkv_gemm<<<2304, 256, 0, stream>>>(xb, WqkvT, bqkv, Qw, Kw, Vt);      // 18432 waves
  scores_gemm<<<6144, 256, 0, stream>>>(Qw, Kw, attn);                  // 128x128 tiles
  mix_softmax<<<B_ * N_, 256, 0, stream>>>(attn, Wl, bl, Ww, bw);       // 8192 blocks
  av_gemm<<<768, 256, 0, stream>>>(attn, Vt, OH);                       // 6144 waves
  proj_gemm<<<768, 256, 0, stream>>>(OH, WpT, bp, out);                 // 6144 waves
}
